// MultiHeadSelfAttention_23811298689986
// MI455X (gfx1250) — matrix operand order repaired and checked by reading, not compiled, not run on hardware
//
#include <hip/hip_runtime.h>
#include <hip/hip_bf16.h>

typedef _Float16 half_t;
typedef __attribute__((ext_vector_type(16))) _Float16 v16h;
typedef __attribute__((ext_vector_type(8)))  _Float16 v8h;
typedef __attribute__((ext_vector_type(8)))  float    v8f;

#define D_MODEL 1024
#define N_HEADS 16
#define D_HEAD  64
#define SEQ     2048
#define LN_EPS  1e-4f

// ---------------------------------------------------------------------------
// WMMA helpers (CDNA5 gfx1250, wave32)
// ---------------------------------------------------------------------------
__device__ __forceinline__ v8f wmma32(v16h a, v16h b, v8f c) {
  // D = A(16x32 f16) * B(32x16 f16) + C(16x16 f32)
  return __builtin_amdgcn_wmma_f32_16x16x32_f16(
      /*neg_a=*/false, a, /*neg_b=*/false, b,
      /*c_mod=*/(short)0, c, /*reuse_a=*/false, /*reuse_b=*/false);
}

// A-matrix fragment (16x32 f16), contiguous-k layout used by this file (matches load_bfrag):
// lane L holds row (L&15); halves 0..15 -> K = (L>>4)*16 + 0..15 (linear).
__device__ __forceinline__ v16h load_afrag(const half_t* p, int ld) {
  const int lane = threadIdx.x & 31;
  const half_t* q = p + (size_t)(lane & 15) * ld + ((lane >> 4) << 4);
  v8h lo = *(const v8h*)(q);
  v8h hi = *(const v8h*)(q + 8);
  v16h r;
#pragma unroll
  for (int i = 0; i < 8; ++i) { r[i] = lo[i]; r[i + 8] = hi[i]; }
  return r;
}

// B-matrix fragment (32x16 f16), same contiguous-k layout as load_afrag:
// lane L holds column (L&15); halves 0..15 -> K = (L>>4)*16 + 0..15 (linear).
// Memory must hold logical B transposed: [N rows x K cols] row-major, so each
// lane reads one contiguous 32-byte run.
__device__ __forceinline__ v16h load_bfrag(const half_t* p, int ld) {
  const int lane = threadIdx.x & 31;
  const half_t* q = p + (size_t)(lane & 15) * ld + ((lane >> 4) << 4);
  return *(const v16h*)q;
}

// ---------------------------------------------------------------------------
// 1) LayerNorm (fp32 in) -> f16 normalized activations
// ---------------------------------------------------------------------------
__global__ __launch_bounds__(256) void ln_kernel(const float* __restrict__ X,
                                                 const float* __restrict__ gw,
                                                 const float* __restrict__ gb,
                                                 half_t* __restrict__ Xn) {
  const int row = blockIdx.x;
  const int tid = threadIdx.x;
  const float* x = X + (size_t)row * D_MODEL;
  float v[4];
  float s = 0.f, ss = 0.f;
#pragma unroll
  for (int i = 0; i < 4; ++i) {
    v[i] = x[tid + i * 256];
    s += v[i];
    ss += v[i] * v[i];
  }
#pragma unroll
  for (int m = 1; m < 32; m <<= 1) {
    s += __shfl_xor(s, m, 32);
    ss += __shfl_xor(ss, m, 32);
  }
  __shared__ float rs[8], rss[8];
  const int wid = tid >> 5;
  if ((tid & 31) == 0) { rs[wid] = s; rss[wid] = ss; }
  __syncthreads();
  if (tid < 32) {
    s = (tid < 8) ? rs[tid] : 0.f;
    ss = (tid < 8) ? rss[tid] : 0.f;
#pragma unroll
    for (int m = 1; m < 8; m <<= 1) {
      s += __shfl_xor(s, m, 32);
      ss += __shfl_xor(ss, m, 32);
    }
    if (tid == 0) { rs[0] = s; rss[0] = ss; }
  }
  __syncthreads();
  const float mean = rs[0] * (1.0f / D_MODEL);
  const float var  = rss[0] * (1.0f / D_MODEL) - mean * mean;
  const float rstd = rsqrtf(var + LN_EPS);
  half_t* xo = Xn + (size_t)row * D_MODEL;
#pragma unroll
  for (int i = 0; i < 4; ++i) {
    const int c = tid + i * 256;
    xo[c] = (half_t)(((v[i] - mean) * rstd) * gw[c] + gb[c]);
  }
}

// ---------------------------------------------------------------------------
// 2) Weight convert + transpose: W fp32 [K x N] -> Wt f16 [N x K]
// ---------------------------------------------------------------------------
__global__ __launch_bounds__(256) void transpose_w(const float* __restrict__ W,
                                                   half_t* __restrict__ Wt) {
  __shared__ float tile[32][33];
  const int bx = blockIdx.x * 32;  // n
  const int by = blockIdx.y * 32;  // k
  for (int i = threadIdx.y; i < 32; i += 8)
    tile[i][threadIdx.x] = W[(size_t)(by + i) * D_MODEL + bx + threadIdx.x];
  __syncthreads();
  for (int i = threadIdx.y; i < 32; i += 8)
    Wt[(size_t)(bx + i) * D_MODEL + by + threadIdx.x] = (half_t)tile[threadIdx.x][i];
}

// ---------------------------------------------------------------------------
// 3) WMMA GEMM: C[M x N] = A[M x K] * Bt[N x K]^T + bias
//    Each wave computes a 32x32 output block (2x2 WMMA tiles, 1:1 load:wmma).
//    mode 0: f16 natural output | mode 1: f16 -> Vt[b][h][dh][s] | mode 2: fp32
// ---------------------------------------------------------------------------
__device__ __forceinline__ void store_tile(v8f c, int rowb, int colb,
                                           const float* __restrict__ bias,
                                           void* __restrict__ out, int N,
                                           int mode, int moff, int n) {
  const float bv = bias[colb + n];
  if (mode == 2) {
    float* o = (float*)out;
#pragma unroll
    for (int r = 0; r < 8; ++r)
      o[(size_t)(rowb + moff + r) * N + colb + n] = c[r] + bv;
  } else if (mode == 0) {
    half_t* o = (half_t*)out;
#pragma unroll
    for (int r = 0; r < 8; ++r)
      o[(size_t)(rowb + moff + r) * N + colb + n] = (half_t)(c[r] + bv);
  } else {  // mode 1: scatter to Vt[b][h][dh][s]
    half_t* o = (half_t*)out;
    const int col = colb + n;
    const int h = col / D_HEAD, dh = col % D_HEAD;
#pragma unroll
    for (int r = 0; r < 8; ++r) {
      const int rr = rowb + moff + r;
      const int b = rr / SEQ, s = rr % SEQ;
      o[((size_t)(b * N_HEADS + h) * D_HEAD + dh) * SEQ + s] = (half_t)(c[r] + bv);
    }
  }
}

__global__ __launch_bounds__(128) void gemm_wmma_kernel(
    const half_t* __restrict__ A, const half_t* __restrict__ Bt,
    const float* __restrict__ bias, void* __restrict__ out,
    int N, int Kd, int mode) {
  const int row0 = blockIdx.x * 32;
  const int col0 = (blockIdx.y * 4 + threadIdx.y) * 32;
  const int lane = threadIdx.x & 31;
  const int n = lane & 15;
  const int moff = (lane >> 4) << 3;

  v8f c00 = {}, c01 = {}, c10 = {}, c11 = {};
  const half_t* pa0 = A + (size_t)row0 * Kd;
  const half_t* pa1 = pa0 + (size_t)16 * Kd;
  const half_t* pb0 = Bt + (size_t)col0 * Kd;
  const half_t* pb1 = pb0 + (size_t)16 * Kd;
  for (int k0 = 0; k0 < Kd; k0 += 32) {
    const v16h a0 = load_afrag(pa0 + k0, Kd);
    const v16h a1 = load_afrag(pa1 + k0, Kd);
    const v16h b0 = load_bfrag(pb0 + k0, Kd);
    const v16h b1 = load_bfrag(pb1 + k0, Kd);
    c00 = wmma32(a0, b0, c00);
    c01 = wmma32(a0, b1, c01);
    c10 = wmma32(a1, b0, c10);
    c11 = wmma32(a1, b1, c11);
  }
  store_tile(c00, row0,      col0,      bias, out, N, mode, moff, n);
  store_tile(c01, row0,      col0 + 16, bias, out, N, mode, moff, n);
  store_tile(c10, row0 + 16, col0,      bias, out, N, mode, moff, n);
  store_tile(c11, row0 + 16, col0 + 16, bias, out, N, mode, moff, n);
}

// ---------------------------------------------------------------------------
// 4) Flash attention: one wave per (b,h, 16-query tile), online softmax over
//    32-key blocks. QK^T and PV via WMMA; P transposed C->A layout via LDS.
// ---------------------------------------------------------------------------
__global__ __launch_bounds__(32) void attn_kernel(const half_t* __restrict__ Qf,
                                                  const half_t* __restrict__ Kf,
                                                  const half_t* __restrict__ Vt,
                                                  half_t* __restrict__ merged) {
  __shared__ __align__(16) half_t pbuf[16 * 40];  // 16 rows, stride 40 halfs
  const int bh = blockIdx.x;
  const int b = bh / N_HEADS, h = bh % N_HEADS;
  const int q0 = blockIdx.y * 16;
  const int lane = threadIdx.x & 31;
  const int n = lane & 15;
  const int moff = (lane >> 4) << 3;
  const float scale = 0.125f;  // 1/sqrt(64)
  const float NEG_INF = -__builtin_huge_valf();

  const half_t* Qbase = Qf + ((size_t)(b * SEQ + q0)) * D_MODEL + h * D_HEAD;
  const v16h aQ0 = load_afrag(Qbase, D_MODEL);
  const v16h aQ1 = load_afrag(Qbase + 32, D_MODEL);

  v8f o0 = {}, o1 = {}, o2 = {}, o3 = {};
  float mrow[8], lrow[8];
#pragma unroll
  for (int r = 0; r < 8; ++r) { mrow[r] = NEG_INF; lrow[r] = 0.f; }

  const int kend = q0 + 16;
  for (int kbase = 0; kbase < kend; kbase += 32) {
    // ---- scores: two 16x16 tiles covering 32 keys ----
    // B = K^T (dh x keys); memory Kf[key][dh] is [N x K] row-major.
    const half_t* KB = Kf + ((size_t)(b * SEQ + kbase)) * D_MODEL + h * D_HEAD;
    const half_t* KB1 = KB + (size_t)16 * D_MODEL;
    v8f s0 = {}, s1 = {};
    s0 = wmma32(aQ0, load_bfrag(KB, D_MODEL), s0);
    s0 = wmma32(aQ1, load_bfrag(KB + 32, D_MODEL), s0);
    s1 = wmma32(aQ0, load_bfrag(KB1, D_MODEL), s1);
    s1 = wmma32(aQ1, load_bfrag(KB1 + 32, D_MODEL), s1);

    // ---- causal mask + online softmax (row = moff+r, col = n) ----
#pragma unroll
    for (int r = 0; r < 8; ++r) {
      const int q = q0 + moff + r;
      float x0 = (kbase + n <= q) ? s0[r] * scale : NEG_INF;
      float x1 = (kbase + 16 + n <= q) ? s1[r] * scale : NEG_INF;
      float rm = fmaxf(x0, x1);
      rm = fmaxf(rm, __shfl_xor(rm, 1, 32));
      rm = fmaxf(rm, __shfl_xor(rm, 2, 32));
      rm = fmaxf(rm, __shfl_xor(rm, 4, 32));
      rm = fmaxf(rm, __shfl_xor(rm, 8, 32));
      const float newm = fmaxf(mrow[r], rm);
      const float alpha = __expf(mrow[r] - newm);
      mrow[r] = newm;
      const float p0 = __expf(x0 - newm);
      const float p1 = __expf(x1 - newm);
      float rsum = p0 + p1;
      rsum += __shfl_xor(rsum, 1, 32);
      rsum += __shfl_xor(rsum, 2, 32);
      rsum += __shfl_xor(rsum, 4, 32);
      rsum += __shfl_xor(rsum, 8, 32);
      lrow[r] = lrow[r] * alpha + rsum;
      o0[r] *= alpha; o1[r] *= alpha; o2[r] *= alpha; o3[r] *= alpha;
      pbuf[(moff + r) * 40 + n]      = (half_t)p0;
      pbuf[(moff + r) * 40 + 16 + n] = (half_t)p1;
    }
    __syncthreads();  // single wave: orders LDS store -> load

    // ---- PV: P (16x32) * V (32 keys x 64 dh); Vt[dh][s] is [N x K] ----
    const v16h aP = load_afrag(pbuf, 40);
    const half_t* VB = Vt + ((size_t)bh * D_HEAD) * SEQ + kbase;
    o0 = wmma32(aP, load_bfrag(VB,                    SEQ), o0);
    o1 = wmma32(aP, load_bfrag(VB + (size_t)16 * SEQ, SEQ), o1);
    o2 = wmma32(aP, load_bfrag(VB + (size_t)32 * SEQ, SEQ), o2);
    o3 = wmma32(aP, load_bfrag(VB + (size_t)48 * SEQ, SEQ), o3);
    __syncthreads();
  }

  half_t* ob = merged + ((size_t)(b * SEQ + q0 + moff)) * D_MODEL + h * D_HEAD + n;
#pragma unroll
  for (int r = 0; r < 8; ++r) {
    const float inv = 1.0f / lrow[r];
    ob[(size_t)r * D_MODEL + 0]  = (half_t)(o0[r] * inv);
    ob[(size_t)r * D_MODEL + 16] = (half_t)(o1[r] * inv);
    ob[(size_t)r * D_MODEL + 32] = (half_t)(o2[r] * inv);
    ob[(size_t)r * D_MODEL + 48] = (half_t)(o3[r] * inv);
  }
}

// ---------------------------------------------------------------------------
// Host launcher
// ---------------------------------------------------------------------------
extern "C" void kernel_launch(void* const* d_in, const int* in_sizes, int n_in,
                              void* d_out, int out_size, void* d_ws, size_t ws_size,
                              hipStream_t stream) {
  (void)n_in; (void)out_size; (void)ws_size;
  const float* X    = (const float*)d_in[0];
  const float* ln_w = (const float*)d_in[1];
  const float* ln_b = (const float*)d_in[2];
  const float* Wq   = (const float*)d_in[3];
  const float* bq   = (const float*)d_in[4];
  const float* Wk   = (const float*)d_in[5];
  const float* bk   = (const float*)d_in[6];
  const float* Wv   = (const float*)d_in[7];
  const float* bv   = (const float*)d_in[8];
  const float* Wo   = (const float*)d_in[9];
  const float* bo   = (const float*)d_in[10];

  const int BS = in_sizes[0] / D_MODEL;  // B * S rows
  const int Bn = BS / SEQ;

  char* ws = (char*)d_ws;
  size_t off = 0;
  auto wsalloc = [&](size_t bytes) -> char* {
    char* p = ws + off;
    off += (bytes + 255) & ~(size_t)255;
    return p;
  };
  const size_t actBytes = (size_t)BS * D_MODEL * sizeof(half_t);
  const size_t wBytes   = (size_t)D_MODEL * D_MODEL * sizeof(half_t);
  half_t* Xn     = (half_t*)wsalloc(actBytes);
  half_t* Wqt    = (half_t*)wsalloc(wBytes);
  half_t* Wkt    = (half_t*)wsalloc(wBytes);
  half_t* Wvt    = (half_t*)wsalloc(wBytes);
  half_t* Wot    = (half_t*)wsalloc(wBytes);
  half_t* Qf     = (half_t*)wsalloc(actBytes);
  half_t* Kf     = (half_t*)wsalloc(actBytes);
  half_t* Vt     = (half_t*)wsalloc(actBytes);  // [B][H][64][S]
  half_t* merged = (half_t*)wsalloc(actBytes);

  // 1) LayerNorm -> f16
  ln_kernel<<<BS, 256, 0, stream>>>(X, ln_w, ln_b, Xn);

  // 2) weight transpose+convert
  {
    dim3 tb(32, 8), tg(D_MODEL / 32, D_MODEL / 32);
    transpose_w<<<tg, tb, 0, stream>>>(Wq, Wqt);
    transpose_w<<<tg, tb, 0, stream>>>(Wk, Wkt);
    transpose_w<<<tg, tb, 0, stream>>>(Wv, Wvt);
    transpose_w<<<tg, tb, 0, stream>>>(Wo, Wot);
  }

  // 3) Q/K/V projections (WMMA, 32x32 per wave)
  dim3 gb(32, 4), gg(BS / 32, D_MODEL / 128);
  gemm_wmma_kernel<<<gg, gb, 0, stream>>>(Xn, Wqt, bq, Qf, D_MODEL, D_MODEL, 0);
  gemm_wmma_kernel<<<gg, gb, 0, stream>>>(Xn, Wkt, bk, Kf, D_MODEL, D_MODEL, 0);
  gemm_wmma_kernel<<<gg, gb, 0, stream>>>(Xn, Wvt, bv, Vt, D_MODEL, D_MODEL, 1);

  // 4) causal flash attention (WMMA)
  attn_kernel<<<dim3(Bn * N_HEADS, SEQ / 16), 32, 0, stream>>>(Qf, Kf, Vt, merged);

  // 5) output projection -> fp32 d_out (WMMA)
  gemm_wmma_kernel<<<gg, gb, 0, stream>>>(merged, Wot, bo, (float*)d_out,
                                          D_MODEL, D_MODEL, 2);
}
